// AttentionBlock_26147760898479
// MI455X (gfx1250) — matrix operand order repaired and checked by reading, not compiled, not run on hardware
//
#include <hip/hip_runtime.h>
#include <hip/hip_bf16.h>

// ---------------------------------------------------------------------------
// AttentionBlock: GroupNorm -> QKV gemm -> MHSA -> proj gemm + residual
// B=16, C=512, H=W=32 (N=1024), heads=8, hd=64, groups=32
// Matmuls via v_wmma_f32_16x16x32_f16; A-tiles staged to LDS via the
// CDNA5 Tensor Data Mover (tensor_load_to_lds + s_wait_tensorcnt).
// ---------------------------------------------------------------------------

typedef __attribute__((ext_vector_type(16))) _Float16     v16h;
typedef __attribute__((ext_vector_type(8)))  float        v8f;
typedef __attribute__((ext_vector_type(4)))  unsigned int u32x4;
typedef __attribute__((ext_vector_type(8)))  int          i32x8;
typedef __attribute__((ext_vector_type(4)))  int          i32x4;

#define Cc   512
#define Nn   1024
#define NH   8
#define HD   64
#define Bb   16

#if __has_builtin(__builtin_amdgcn_tensor_load_to_lds)
#define AB_HAVE_TDM 1
#endif

__device__ __forceinline__ v8f wmma16(v16h a, v16h b, v8f c) {
  // (neg_a, A, neg_b, B, c_mod, C, reuse_a, reuse_b)
  return __builtin_amdgcn_wmma_f32_16x16x32_f16(false, a, false, b, (short)0, c,
                                                false, false);
}

// A-matrix fragment (16x32 f16).  ISA layout: lanes 0-15 hold M=lane,
// K={0..7,16..23}; lanes 16-31 hold M=lane-16, K={8..15,24..31}.
__device__ __forceinline__ v16h load_a16(const _Float16* __restrict__ A, int lda,
                                         int kbase, int lane) {
  const int m  = lane & 15;
  const int ks = kbase + ((lane >> 4) << 3);
  const _Float16* p = A + m * lda + ks;
  v16h a;
#pragma unroll
  for (int i = 0; i < 8; ++i) { a[i] = p[i]; a[8 + i] = p[16 + i]; }
  return a;
}

// Same, but A lives in LDS as f32 (post-softmax probabilities).
__device__ __forceinline__ v16h load_a16_lds(const float* __restrict__ S,
                                             int kbase, int lane) {
  const int m  = lane & 15;
  const int ks = kbase + ((lane >> 4) << 3);
  const float* p = S + m * Nn + ks;
  v16h a;
#pragma unroll
  for (int i = 0; i < 8; ++i) {
    a[i]     = (_Float16)p[i];
    a[8 + i] = (_Float16)p[16 + i];
  }
  return a;
}

// B-matrix fragment (32x16 f16) supplied as BT[n][k] (K contiguous per column).
__device__ __forceinline__ v16h load_bT16(const _Float16* __restrict__ BT, int ldb,
                                          int kbase, int lane) {
  const int n  = lane & 15;
  const int ks = kbase + ((lane >> 4) << 3);
  const _Float16* p = BT + n * ldb + ks;
  v16h b;
#pragma unroll
  for (int i = 0; i < 8; ++i) { b[i] = p[i]; b[8 + i] = p[16 + i]; }
  return b;
}

// wave32 xor-shuffle via ds_swizzle (group-of-32 mode: and=0x1f, or=0, xor=XM)
template <int XM>
__device__ __forceinline__ float swz_xor(float v) {
  return __int_as_float(
      __builtin_amdgcn_ds_swizzle(__float_as_int(v), 0x1f | (XM << 10)));
}

// ---------------------------------------------------------------------------
// TDM: async-load a 2D f16 tile (rows x row_elems, row stride in elements)
// from global memory into LDS.  D# bit-packing per CDNA5 ISA ch.8.
// ---------------------------------------------------------------------------
__device__ __forceinline__ void tdm_load_2d_f16(const _Float16* gptr,
                                                void* lds_dst, int rows,
                                                int row_elems,
                                                int row_stride_elems) {
#if defined(AB_HAVE_TDM)
  const unsigned long long ga = (unsigned long long)(size_t)gptr;
  const unsigned ldsoff = (unsigned)(size_t)lds_dst;  // LDS byte offset

  u32x4 g0;
  g0[0] = 1u;                                          // count=1, user D#
  g0[1] = ldsoff;                                      // lds_addr
  g0[2] = (unsigned)(ga & 0xFFFFFFFFull);              // global_addr[31:0]
  g0[3] = (unsigned)((ga >> 32) & 0x01FFFFFFull) | (2u << 30);  // [56:32]|type=2

  i32x8 g1;
  g1[0] = (int)(1u << 16);                             // data_size=1 (2 bytes)
  g1[1] = (int)(((unsigned)row_elems & 0xFFFFu) << 16);       // tensor_dim0 lo
  g1[2] = (int)((((unsigned)row_elems >> 16) & 0xFFFFu) |
                (((unsigned)rows & 0xFFFFu) << 16));          // dim0 hi|dim1 lo
  g1[3] = (int)(((unsigned)row_elems & 0xFFFFu) << 16);       // tile_dim0
  g1[4] = (int)((unsigned)rows & 0xFFFFu);                    // tile_dim1
  g1[5] = row_stride_elems;                                   // dim0_stride lo
  g1[6] = 0;
  g1[7] = 0;

  i32x4 z4 = {0, 0, 0, 0};
#if defined(__clang_major__) && (__clang_major__ >= 23)
  i32x8 z8 = {0, 0, 0, 0, 0, 0, 0, 0};
  __builtin_amdgcn_tensor_load_to_lds(g0, g1, z4, z4, z8, 0);
#else
  __builtin_amdgcn_tensor_load_to_lds(g0, g1, z4, z4, 0);
#endif
#endif
}

// Cooperative stage of one 16x512 f16 tile into LDS (TDM when available).
__device__ __forceinline__ void stage_a_tile(const _Float16* __restrict__ A,
                                             _Float16* sA) {
#if defined(AB_HAVE_TDM)
  if ((threadIdx.x >> 5) == 0) {
    tdm_load_2d_f16(A, sA, 16, Cc, Cc);
    __builtin_amdgcn_s_wait_tensorcnt(0);
  }
  __syncthreads();
#else
  // fallback: plain cooperative copy (16 KB = 1024 x uint4)
  const uint4* src = (const uint4*)A;
  uint4* dst = (uint4*)sA;
  for (int i = threadIdx.x; i < (16 * Cc) / 8; i += 256) dst[i] = src[i];
  __syncthreads();
#endif
}

// ---------------------------------------------------------------------------
// Kernel 1: f32 -> f16 weight conversion (qkv_w: 1536x512, proj_w: 512x512)
// ---------------------------------------------------------------------------
__global__ __launch_bounds__(256) void ab_wconv_kernel(
    const float* __restrict__ qkv_w, const float* __restrict__ proj_w,
    _Float16* __restrict__ wq16, _Float16* __restrict__ wp16) {
  const int i = blockIdx.x * 256 + threadIdx.x;
  const int NQ = 3 * Cc * Cc;  // 786432
  if (i < NQ) {
    wq16[i] = (_Float16)qkv_w[i];
  } else {
    const int j = i - NQ;  // < 262144
    wp16[j] = (_Float16)proj_w[j];
  }
}

// ---------------------------------------------------------------------------
// Kernel 2: GroupNorm(32, 512); writes hn as f16 in (B*N, C) row-major.
// ---------------------------------------------------------------------------
__global__ __launch_bounds__(256) void ab_groupnorm_kernel(
    const float* __restrict__ x, const float* __restrict__ gw,
    const float* __restrict__ gb, _Float16* __restrict__ hn) {
  const int b  = blockIdx.x >> 5;
  const int g  = blockIdx.x & 31;
  const int cg = g * 16;
  const int tid = threadIdx.x;

  float s = 0.f, ss = 0.f;
  for (int i = tid; i < 16 * Nn; i += 256) {
    const int c = i >> 10;
    const int p = i & (Nn - 1);
    const float v = x[((b * Cc + cg + c) << 10) + p];
    s += v;
    ss += v * v;
  }
  __shared__ float r1[256], r2[256];
  __shared__ float stats[2];
  r1[tid] = s; r2[tid] = ss;
  __syncthreads();
  for (int o = 128; o > 0; o >>= 1) {
    if (tid < o) { r1[tid] += r1[tid + o]; r2[tid] += r2[tid + o]; }
    __syncthreads();
  }
  if (tid == 0) {
    const float mean = r1[0] * (1.0f / 16384.0f);
    const float var  = r2[0] * (1.0f / 16384.0f) - mean * mean;
    stats[0] = mean;
    stats[1] = rsqrtf(var + 1e-5f);
  }
  __syncthreads();
  const float mean = stats[0];
  const float rinv = stats[1];

  for (int i = tid; i < 16 * Nn; i += 256) {
    const int c  = i >> 10;
    const int p  = i & (Nn - 1);
    const int cc = cg + c;
    const float v = x[((b * Cc + cc) << 10) + p];
    const float hv = (v - mean) * rinv * gw[cc] + gb[cc];
    hn[((b << 10) + p) * Cc + cc] = (_Float16)hv;
  }
}

// ---------------------------------------------------------------------------
// Kernel 3: QKV GEMM (M=16384, K=512, Nout=1536).
// A tile (16x512 f16, shared by all 8 waves) staged to LDS via TDM.
// Each wave computes a 16x32 output slab (2 WMMA tiles, A reused).
// grid (1024, 6), 8 waves/block.
// ---------------------------------------------------------------------------
__global__ __launch_bounds__(256) void ab_qkv_gemm_kernel(
    const _Float16* __restrict__ hn, const _Float16* __restrict__ wq,
    const float* __restrict__ qkv_b, _Float16* __restrict__ q16,
    _Float16* __restrict__ k16, _Float16* __restrict__ vt16) {
  __shared__ _Float16 sA[16 * Cc];  // 16 KB

  const int lane  = threadIdx.x & 31;
  const int wave  = threadIdx.x >> 5;
  const int mbase = blockIdx.x * 16;
  const int nbase = (blockIdx.y * 8 + wave) * 32;

  stage_a_tile(hn + (size_t)mbase * Cc, sA);

  const _Float16* BT0 = wq + (size_t)nbase * Cc;
  const _Float16* BT1 = BT0 + (size_t)16 * Cc;

  v8f acc0 = {0.f, 0.f, 0.f, 0.f, 0.f, 0.f, 0.f, 0.f};
  v8f acc1 = {0.f, 0.f, 0.f, 0.f, 0.f, 0.f, 0.f, 0.f};
#pragma unroll 4
  for (int kk = 0; kk < Cc; kk += 32) {
    if (kk + 64 < Cc) __builtin_prefetch(BT0 + (lane & 15) * Cc + kk + 64, 0, 1);
    v16h a = load_a16(sA, Cc, kk, lane);
    acc0 = wmma16(a, load_bT16(BT0, Cc, kk, lane), acc0);
    acc1 = wmma16(a, load_bT16(BT1, Cc, kk, lane), acc1);
  }

  const int n  = lane & 15;
  const int mb = (lane >> 4) << 3;
#pragma unroll
  for (int t = 0; t < 2; ++t) {
    const v8f acc = t ? acc1 : acc0;
    const int o  = nbase + t * 16 + n;
    const int which = o >> 9;  // 0=Q 1=K 2=V (uniform per wave per tile)
    const int c = o & (Cc - 1);
    const int h = c >> 6;
    const int d = c & (HD - 1);
    const float bias = qkv_b[o];
#pragma unroll
    for (int r = 0; r < 8; ++r) {
      const int bp = mbase + mb + r;
      const int b_ = bp >> 10;
      const int p  = bp & (Nn - 1);
      const int bh = b_ * NH + h;
      const float v = acc[r] + bias;
      if (which == 0)
        q16[((size_t)bh * Nn + p) * HD + d] = (_Float16)(v * 0.125f);
      else if (which == 1)
        k16[((size_t)bh * Nn + p) * HD + d] = (_Float16)v;
      else
        vt16[((size_t)bh * HD + d) * Nn + p] = (_Float16)v;
    }
  }
}

// ---------------------------------------------------------------------------
// Kernel 4: attention for one (b, h, 16-query-row tile).  4 waves/block:
// score phase splits the 64 key tiles, softmax splits rows (ds_swizzle
// reductions), PV phase splits the 4 output d-tiles.  S = 64 KB LDS.
// ---------------------------------------------------------------------------
__global__ __launch_bounds__(128) void ab_attn_kernel(
    const _Float16* __restrict__ q16, const _Float16* __restrict__ k16,
    const _Float16* __restrict__ vt16, _Float16* __restrict__ o16) {
  __shared__ float S[16 * Nn];  // 65536 bytes

  const int bh   = blockIdx.x >> 6;  // b*8+h
  const int qt   = blockIdx.x & 63;  // 16-row query tile
  const int wave = threadIdx.x >> 5;
  const int lane = threadIdx.x & 31;

  const _Float16* qb = q16 + ((size_t)bh * Nn + qt * 16) * HD;
  const _Float16* kb = k16 + (size_t)bh * Nn * HD;
  const _Float16* vb = vt16 + (size_t)bh * HD * Nn;

  const v16h qa0 = load_a16(qb, HD, 0, lane);
  const v16h qa1 = load_a16(qb, HD, 32, lane);

  // ---- scores: S[m][j*16+n] = sum_d Q[m][d] * K[j*16+n][d] -----------------
  for (int jj = 0; jj < 16; ++jj) {
    const int j = wave * 16 + jj;
    const _Float16* kt = kb + (size_t)j * 16 * HD;
    v8f acc = {0.f, 0.f, 0.f, 0.f, 0.f, 0.f, 0.f, 0.f};
    acc = wmma16(qa0, load_bT16(kt, HD, 0, lane), acc);
    acc = wmma16(qa1, load_bT16(kt, HD, 32, lane), acc);
    const int n  = lane & 15;
    const int mb = (lane >> 4) << 3;
    float* sc = S + j * 16 + n;
#pragma unroll
    for (int r = 0; r < 8; ++r) sc[(mb + r) * Nn] = acc[r];
  }
  __syncthreads();

  // ---- softmax over 1024 keys per row: 8 lanes/row, 128 cols/lane ----------
  {
    const int row = wave * 4 + (lane >> 3);
    const int c0  = (lane & 7) * 128;
    float* Sr = S + row * Nn + c0;

    float mx = -1e30f;
#pragma unroll 8
    for (int i = 0; i < 128; ++i) mx = fmaxf(mx, Sr[i]);
    mx = fmaxf(mx, swz_xor<1>(mx));
    mx = fmaxf(mx, swz_xor<2>(mx));
    mx = fmaxf(mx, swz_xor<4>(mx));

    float sum = 0.f;
#pragma unroll 4
    for (int i = 0; i < 128; ++i) {
      const float e = __expf(Sr[i] - mx);
      Sr[i] = e;
      sum += e;
    }
    sum += swz_xor<1>(sum);
    sum += swz_xor<2>(sum);
    sum += swz_xor<4>(sum);
    const float inv = 1.0f / sum;
#pragma unroll 8
    for (int i = 0; i < 128; ++i) Sr[i] *= inv;
  }
  __syncthreads();

  // ---- O tile = P (16x1024, LDS) x V (1024x64): wave w -> d cols 16w..+15 --
  v8f oacc = {0.f, 0.f, 0.f, 0.f, 0.f, 0.f, 0.f, 0.f};
  const _Float16* vbt = vb + (size_t)(wave * 16) * Nn;
  for (int kk = 0; kk < Nn; kk += 32) {
    v16h pa = load_a16_lds(S, kk, lane);
    v16h pb = load_bT16(vbt, Nn, kk, lane);
    oacc = wmma16(pa, pb, oacc);
  }

  const int b_ = bh >> 3;
  const int h_ = bh & 7;
  _Float16* ob =
      o16 + ((size_t)(b_ * Nn + qt * 16)) * Cc + h_ * HD + wave * 16;
  const int n  = lane & 15;
  const int mb = (lane >> 4) << 3;
#pragma unroll
  for (int r = 0; r < 8; ++r) ob[(mb + r) * Cc + n] = (_Float16)oacc[r];
}

// ---------------------------------------------------------------------------
// Kernel 5: proj GEMM (M=16384, K=512, N=512) + bias + residual, f32 out.
// A tile via TDM -> LDS; wave computes 16x32.  grid (1024, 2).
// ---------------------------------------------------------------------------
__global__ __launch_bounds__(256) void ab_proj_gemm_kernel(
    const _Float16* __restrict__ o16, const _Float16* __restrict__ wp,
    const float* __restrict__ proj_b, const float* __restrict__ x,
    float* __restrict__ out) {
  __shared__ _Float16 sA[16 * Cc];  // 16 KB

  const int lane  = threadIdx.x & 31;
  const int wave  = threadIdx.x >> 5;
  const int mbase = blockIdx.x * 16;
  const int nbase = (blockIdx.y * 8 + wave) * 32;

  stage_a_tile(o16 + (size_t)mbase * Cc, sA);

  const _Float16* BT0 = wp + (size_t)nbase * Cc;
  const _Float16* BT1 = BT0 + (size_t)16 * Cc;

  v8f acc0 = {0.f, 0.f, 0.f, 0.f, 0.f, 0.f, 0.f, 0.f};
  v8f acc1 = {0.f, 0.f, 0.f, 0.f, 0.f, 0.f, 0.f, 0.f};
#pragma unroll 4
  for (int kk = 0; kk < Cc; kk += 32) {
    if (kk + 64 < Cc) __builtin_prefetch(BT0 + (lane & 15) * Cc + kk + 64, 0, 1);
    v16h a = load_a16(sA, Cc, kk, lane);
    acc0 = wmma16(a, load_bT16(BT0, Cc, kk, lane), acc0);
    acc1 = wmma16(a, load_bT16(BT1, Cc, kk, lane), acc1);
  }

  const int n  = lane & 15;
  const int mb = (lane >> 4) << 3;
#pragma unroll
  for (int t = 0; t < 2; ++t) {
    const v8f acc = t ? acc1 : acc0;
    const int o = nbase + t * 16 + n;
    const float bias = proj_b[o];
#pragma unroll
    for (int r = 0; r < 8; ++r) {
      const int bp = mbase + mb + r;
      const int b_ = bp >> 10;
      const int p  = bp & (Nn - 1);
      const size_t gi = (((size_t)b_ * Cc + o) << 10) + p;
      out[gi] = acc[r] + bias + x[gi];
    }
  }
}

// ---------------------------------------------------------------------------
extern "C" void kernel_launch(void* const* d_in, const int* in_sizes, int n_in,
                              void* d_out, int out_size, void* d_ws,
                              size_t ws_size, hipStream_t stream) {
  const float* x      = (const float*)d_in[0];
  const float* norm_w = (const float*)d_in[1];
  const float* norm_b = (const float*)d_in[2];
  const float* qkv_w  = (const float*)d_in[3];
  const float* qkv_b  = (const float*)d_in[4];
  const float* proj_w = (const float*)d_in[5];
  const float* proj_b = (const float*)d_in[6];
  float* out = (float*)d_out;

  char* ws = (char*)d_ws;
  const size_t MB16 = (size_t)16 << 20;  // each big f16 tensor is exactly 16 MB
  _Float16* hn16 = (_Float16*)(ws + 0 * MB16);
  _Float16* q16  = (_Float16*)(ws + 1 * MB16);
  _Float16* k16  = (_Float16*)(ws + 2 * MB16);
  _Float16* vt16 = (_Float16*)(ws + 3 * MB16);
  _Float16* o16  = (_Float16*)(ws + 4 * MB16);
  _Float16* wq16 = (_Float16*)(ws + 5 * MB16);
  _Float16* wp16 = (_Float16*)(ws + 5 * MB16 + (size_t)(3 * Cc * Cc) * 2);

  ab_wconv_kernel<<<4096, 256, 0, stream>>>(qkv_w, proj_w, wq16, wp16);
  ab_groupnorm_kernel<<<Bb * 32, 256, 0, stream>>>(x, norm_w, norm_b, hn16);
  ab_qkv_gemm_kernel<<<dim3(1024, 6), 256, 0, stream>>>(hn16, wq16, qkv_b, q16,
                                                        k16, vt16);
  ab_attn_kernel<<<Bb * NH * (Nn / 16), 128, 0, stream>>>(q16, k16, vt16, o16);
  ab_proj_gemm_kernel<<<dim3(1024, 2), 256, 0, stream>>>(o16, wp16, proj_b, x,
                                                         out);
}
